// Net_40321152975542
// MI455X (gfx1250) — matrix operand order repaired and checked by reading, not compiled, not run on hardware
//
#include <hip/hip_runtime.h>

// CDNA5 (gfx1250) wave32 WMMA residual-MLP evaluator.
// D(16x16,f32) = A(16x32,f16: weights+bias) x B(32x16,f16: activations+const1) + C
// Residual adds ride in the WMMA C accumulator.
typedef __attribute__((ext_vector_type(16))) _Float16 v16h;
typedef __attribute__((ext_vector_type(2)))  __fp16   v2fp16;
typedef __attribute__((ext_vector_type(8)))  float    v8f;

union V16H {
  v16h     v;
  _Float16 f[16];
  unsigned u[8];
};

__device__ __forceinline__ v8f wmma_f16(v16h a, v16h b, v8f c) {
  // (neg_a, A, neg_b, B, c_mod, C, reuse_a, reuse_b)
  return __builtin_amdgcn_wmma_f32_16x16x32_f16(false, a, false, b, (short)0, c,
                                                false, false);
}

// Packed f32x2 -> f16x2 convert (v_cvt_pk_rtz_f16_f32).
__device__ __forceinline__ unsigned pk2(float a, float b) {
  union { v2fp16 h; _Float16 f[2]; unsigned u; } c;
#if __has_builtin(__builtin_amdgcn_cvt_pkrtz)
  c.h = __builtin_amdgcn_cvt_pkrtz(a, b);
#else
  c.f[0] = (_Float16)a;
  c.f[1] = (_Float16)b;
#endif
  return c.u;
}

// relu; med3(x, 0, +inf) per element, 8 per call.
__device__ __forceinline__ v8f vrelu(v8f a) {
#pragma unroll
  for (int j = 0; j < 8; ++j)
    a[j] = __builtin_amdgcn_fmed3f(a[j], 0.0f, __builtin_inff());
  return a;
}

// Turn post-activation f32 in C/D layout (8 floats/lane) into the B operand for
// the next layer. D layout: lanes 0-15 hold features 0..7 of sample=lane,
// lanes 16-31 hold features 8..15 of sample=lane-16.
// B layout (32x16 f16): lanes 0-15 hold K=0..15 of column N=lane (all 16
// features of the sample), lanes 16-31 hold K=16..31 of column N=lane-16.
// K=16 (upper lanes, element 0) carries the constant 1.0 that multiplies the
// bias column of A.
__device__ __forceinline__ v16h act_to_B(v8f h, bool lo) {
  V16H b;
#if __has_builtin(__builtin_amdgcn_permlane16_swap)
  // One v_permlane16_swap_b32 per dword pair produces BOTH b.u[r] and
  // b.u[4+r]: it swaps vdst.lanes16-31 with vsrc.lanes0-15, so
  //   new_vdst = {p.row0, c.row0}  -> own pack low, constant high
  //   new_vsrc = {p.row1, c.row1}  -> partner pack low, zero high
#pragma unroll
  for (int r = 0; r < 4; ++r) {
    unsigned p = pk2(h[2 * r], h[2 * r + 1]);
    unsigned c = (r == 0) ? (lo ? 0x3c00u : 0u) : 0u;  // row0=0x3c00, row1=0
    auto pr    = __builtin_amdgcn_permlane16_swap(p, c, false, false);
    b.u[r]     = pr[0];
    b.u[4 + r] = pr[1];
  }
#else
  unsigned p[4];
#pragma unroll
  for (int r = 0; r < 4; ++r) p[r] = pk2(h[2 * r], h[2 * r + 1]);
#pragma unroll
  for (int r = 0; r < 4; ++r) {
    unsigned q = __shfl_xor(p[r], 16, 32);             // half-wave swap
    b.u[r]     = lo ? p[r] : (r == 0 ? 0x3c00u : 0u);  // 0x3c00 = f16 1.0
    b.u[4 + r] = lo ? q : 0u;
  }
#endif
  return b.v;
}

// fc1 input: x is (B,4) f32 row-major. All lanes load (upper lanes read the
// same float4 as their partner -> same cachelines, result discarded by select).
__device__ __forceinline__ v16h load_x_B(const float* __restrict__ x, int tile,
                                         bool lo) {
  const int row = (int)(threadIdx.x & 15u);
  const float4 xv =
      *reinterpret_cast<const float4*>(x + (size_t)(tile * 16 + row) * 4);
  const unsigned p0 = pk2(xv.x, xv.y);
  const unsigned p1 = pk2(xv.z, xv.w);
  V16H b;
  b.u[0] = lo ? p0 : 0x3c00u;
  b.u[1] = lo ? p1 : 0u;
#pragma unroll
  for (int r = 2; r < 8; ++r) b.u[r] = 0u;
  return b.v;
}

__global__ __launch_bounds__(256) void resmlp_wmma_kernel(
    const float* __restrict__ x,
    const float* __restrict__ W1, const float* __restrict__ b1,
    const float* __restrict__ Wh, const float* __restrict__ bh,
    const float* __restrict__ Wo, const float* __restrict__ bo,
    float* __restrict__ out, int numTiles) {
  const int  lane = (int)(threadIdx.x & 31u);
  const int  row  = lane & 15;
  const bool lo   = lane < 16;

  // ---- Build 12 A-matrix operands (f16 weights, bias folded in at K=16) ----
  // A layout (16x32 f16): lanes 0-15 hold row M=lane, K=0..15 in elems 0..15;
  // lanes 16-31 hold row M=lane-16, K=16..31 in elems 0..15.
  v16h A[12];
  {  // fc1: W1 is (16,4); b1 is (16)
    V16H a;
#pragma unroll
    for (int r = 0; r < 8; ++r) a.u[r] = 0u;
    if (lo) {
      const float* w = W1 + row * 4;
#pragma unroll
      for (int j = 0; j < 4; ++j) a.f[j] = (_Float16)w[j];
    } else {
      a.f[0] = (_Float16)b1[row];  // K=16 -> bias
    }
    A[0] = a.v;
  }
#pragma unroll
  for (int i = 0; i < 10; ++i) {  // hidden: Wh[i] is (16,16); bh[i] is (16)
    V16H a;
#pragma unroll
    for (int r = 0; r < 8; ++r) a.u[r] = 0u;
    const float* w = Wh + i * 256 + row * 16;
    if (lo) {
#pragma unroll
      for (int j = 0; j < 16; ++j) a.f[j] = (_Float16)w[j];
    } else {
      a.f[0] = (_Float16)bh[i * 16 + row];
    }
    A[1 + i] = a.v;
  }
  {  // head: Wo is (2,16); bo is (2) -> rows 0..1, rest zero
    V16H a;
#pragma unroll
    for (int r = 0; r < 8; ++r) a.u[r] = 0u;
    if (row < 2) {
      const float* w = Wo + row * 16;
      if (lo) {
#pragma unroll
        for (int j = 0; j < 16; ++j) a.f[j] = (_Float16)w[j];
      } else {
        a.f[0] = (_Float16)bo[row];
      }
    }
    A[11] = a.v;
  }

  const int wid    = (int)((blockIdx.x * blockDim.x + threadIdx.x) >> 5);
  const int nWaves = (int)((gridDim.x * blockDim.x) >> 5);
  const v8f z = {};

  // Two independent 16-sample tiles per iteration; WMMAs for both tiles are
  // issued back-to-back so each covers the other's D->VALU hazard window.
  for (int tile = wid * 2; tile < numTiles; tile += nWaves * 2) {
    int t[2];
    t[0] = tile;
    t[1] = (tile + 1 < numTiles) ? (tile + 1) : tile;

    v16h b[2];
    v8f  d[2], h[2], res[2];
#pragma unroll
    for (int p = 0; p < 2; ++p) b[p] = load_x_B(x, t[p], lo);
#pragma unroll
    for (int p = 0; p < 2; ++p) d[p] = wmma_f16(A[0], b[p], z);  // fc1 + bias
#pragma unroll
    for (int p = 0; p < 2; ++p) {
      h[p]   = vrelu(d[p]);
      res[p] = h[p];
    }
#pragma unroll
    for (int p = 0; p < 2; ++p) b[p] = act_to_B(h[p], lo);

#pragma unroll
    for (int i = 0; i < 5; ++i) {
      // fcA: h = relu(Wa h + ba)
#pragma unroll
      for (int p = 0; p < 2; ++p) d[p] = wmma_f16(A[1 + 2 * i], b[p], z);
#pragma unroll
      for (int p = 0; p < 2; ++p) h[p] = vrelu(d[p]);
#pragma unroll
      for (int p = 0; p < 2; ++p) b[p] = act_to_B(h[p], lo);
      // fcB: h = relu(Wb h + bb + res); residual rides in the C accumulator
#pragma unroll
      for (int p = 0; p < 2; ++p) d[p] = wmma_f16(A[2 + 2 * i], b[p], res[p]);
#pragma unroll
      for (int p = 0; p < 2; ++p) {
        h[p]   = vrelu(d[p]);
        res[p] = h[p];
      }
#pragma unroll
      for (int p = 0; p < 2; ++p) b[p] = act_to_B(h[p], lo);
    }

    // head: out = Wo h + bo (rows 0,1 of D)
#pragma unroll
    for (int p = 0; p < 2; ++p) d[p] = wmma_f16(A[11], b[p], z);
#pragma unroll
    for (int p = 0; p < 2; ++p) {
      if (lo) {
        float2 o;
        o.x = d[p][0];
        o.y = d[p][1];
        *reinterpret_cast<float2*>(out + (size_t)(t[p] * 16 + row) * 2) = o;
      }
    }
  }
}

extern "C" void kernel_launch(void* const* d_in, const int* in_sizes, int n_in,
                              void* d_out, int out_size, void* d_ws,
                              size_t ws_size, hipStream_t stream) {
  const float* x  = (const float*)d_in[0];
  const float* W1 = (const float*)d_in[1];
  const float* b1 = (const float*)d_in[2];
  const float* Wh = (const float*)d_in[3];
  const float* bh = (const float*)d_in[4];
  const float* Wo = (const float*)d_in[5];
  const float* bo = (const float*)d_in[6];
  float* out = (float*)d_out;

  const int Bsz      = in_sizes[0] / 4;  // N_STATES = 4
  const int numTiles = Bsz / 16;         // B is a multiple of 16

  int blocks = (numTiles + 15) / 16;  // 8 waves/block * 2 tiles/wave/iter
  if (blocks > 1024) blocks = 1024;
  if (blocks < 1) blocks = 1;
  resmlp_wmma_kernel<<<blocks, 256, 0, stream>>>(x, W1, b1, Wh, bh, Wo, bo, out,
                                                 numTiles);
}
